// Attention_51900384805207
// MI455X (gfx1250) — matrix operand order repaired and checked by reading, not compiled, not run on hardware
//
#include <hip/hip_runtime.h>

typedef __attribute__((ext_vector_type(16))) _Float16 v16h;
typedef __attribute__((ext_vector_type(8)))  _Float16 v8h;
typedef __attribute__((ext_vector_type(8)))  float    v8f;

#define N_TOK  2304
#define C_IN   256
#define HEADS  8
#define DHEAD  32
#define HIDDEN 256
#define O3     768
#define BATCH  4
#define QSCALE 0.17677669529663687f   // 32^-0.5

__device__ __forceinline__ v16h combine(v8h lo, v8h hi) {
  v16h r;
  #pragma unroll
  for (int i = 0; i < 8; i++) { r[i] = lo[i]; r[i + 8] = hi[i]; }
  return r;
}

// ---------------------------------------------------------------------------
// Kernel 0a: convert both weight matrices to f16 once (removes per-tile cvts).
// ---------------------------------------------------------------------------
__global__ __launch_bounds__(256) void cvt_w_k(
    const float* __restrict__ wq, const float* __restrict__ wo,
    _Float16* __restrict__ wqh, _Float16* __restrict__ woh)
{
  int i = blockIdx.x * 256 + threadIdx.x;
  if (i < O3 * C_IN)     wqh[i] = (_Float16)wq[i];
  if (i < C_IN * HIDDEN) woh[i] = (_Float16)wo[i];
}

// ---------------------------------------------------------------------------
// Kernel 0b: xT[b][n][c] = (f16) x[b][c][n]  via 32x32 LDS tiles, so the QKV
// GEMM B-fragments become contiguous 16B loads instead of 16 strided b32s.
// ---------------------------------------------------------------------------
__global__ __launch_bounds__(256) void xpose_k(
    const float* __restrict__ x, _Float16* __restrict__ xT)
{
  __shared__ _Float16 tile[32][33];
  int bid = blockIdx.x;
  int nt = bid % (N_TOK / 32);
  int ct = (bid / (N_TOK / 32)) % (C_IN / 32);
  int b  = bid / ((N_TOK / 32) * (C_IN / 32));
  int c0 = ct * 32, n0 = nt * 32;
  int tx = threadIdx.x & 31, ty = threadIdx.x >> 5;
  #pragma unroll
  for (int i = 0; i < 4; i++) {
    int c = ty + i * 8;
    tile[c][tx] = (_Float16)x[((size_t)b * C_IN + c0 + c) * N_TOK + n0 + tx];
  }
  __syncthreads();
  #pragma unroll
  for (int i = 0; i < 4; i++) {
    int nr = ty + i * 8;
    xT[((size_t)b * N_TOK + n0 + nr) * C_IN + c0 + tx] = tile[tx][nr];
  }
}

// ---------------------------------------------------------------------------
// Kernel 1: qkv = w_qkv @ x (768 x 2304, K=256 per batch), all-f16 fragments.
// Q,K stored [b][h][n][32]; V transposed [b][h][32][n]. Q pre-scaled.
// ---------------------------------------------------------------------------
__global__ __launch_bounds__(256) void qkv_gemm_k(
    const _Float16* __restrict__ xT,    // [B][N][C]
    const _Float16* __restrict__ wqkv,  // [768][256] f16
    _Float16* __restrict__ Q,
    _Float16* __restrict__ Kd,
    _Float16* __restrict__ Vt)
{
  int wid  = blockIdx.x * 8 + (threadIdx.x >> 5);
  int lane = threadIdx.x & 31;
  int jt = wid % (N_TOK / 16);
  int mt = (wid / (N_TOK / 16)) % (O3 / 16);
  int b  = wid / ((N_TOK / 16) * (O3 / 16));
  int j0 = jt * 16, m0 = mt * 16;
  int lj = lane & 15, sel = lane >> 4;
  int jcol = j0 + lj;

  const _Float16* arow = wqkv + (size_t)(m0 + lj) * C_IN;
  const _Float16* brow = xT + ((size_t)b * N_TOK + jcol) * C_IN;

  v8f acc = {};
  #pragma unroll
  for (int kc = 0; kc < 8; kc++) {
    int k0 = kc * 32;
    const _Float16* pa = arow + k0 + sel * 16;
    v16h a = combine(*(const v8h*)pa, *(const v8h*)(pa + 8));
    const _Float16* pb = brow + k0 + sel * 16;
    v16h bf = combine(*(const v8h*)pb, *(const v8h*)(pb + 8));
    acc = __builtin_amdgcn_wmma_f32_16x16x32_f16(false, a, false, bf,
                                                 (short)0, acc, false, false);
  }

  int msel8 = sel * 8;
  if (m0 < 256) {
    #pragma unroll
    for (int r = 0; r < 8; r++) {
      int o = m0 + r + msel8;
      int h = o >> 5, d = o & 31;
      Q[(((size_t)b * HEADS + h) * N_TOK + jcol) * DHEAD + d] =
          (_Float16)(acc[r] * QSCALE);
    }
  } else if (m0 < 512) {
    #pragma unroll
    for (int r = 0; r < 8; r++) {
      int o = m0 - 256 + r + msel8;
      int h = o >> 5, d = o & 31;
      Kd[(((size_t)b * HEADS + h) * N_TOK + jcol) * DHEAD + d] = (_Float16)acc[r];
    }
  } else {
    #pragma unroll
    for (int r = 0; r < 8; r++) {
      int o = m0 - 512 + r + msel8;
      int h = o >> 5, d = o & 31;
      Vt[(((size_t)b * HEADS + h) * DHEAD + d) * N_TOK + jcol] = (_Float16)acc[r];
    }
  }
}

// ---------------------------------------------------------------------------
// Kernel 2: flash attention. 8 waves/block cooperate on one (b,h): each wave
// owns one 16-query tile; each 32-key K/V chunk (4KB) is async-staged into
// LDS ONCE per block (global_load_async_to_lds_b128, double buffered,
// s_wait_asynccnt + barrier) and shared by all 8 waves -> 8x less K/V traffic.
// ---------------------------------------------------------------------------
__global__ __launch_bounds__(256) void attn_k(
    const _Float16* __restrict__ Q,
    const _Float16* __restrict__ Kd,
    const _Float16* __restrict__ Vt,
    const float* __restrict__ bias,   // [H][N][N]
    _Float16* __restrict__ AO)        // [B][N][HIDDEN]
{
  __shared__ __align__(16) _Float16 KVs[2][2][1024];  // [buf][K/V][32x32] 8KB
  __shared__ __align__(16) _Float16 Plds[8][512];     // per-wave P scratch 8KB
  int t = threadIdx.x;
  int warp = t >> 5, lane = t & 31;
  int bid = blockIdx.x;
  int b  = bid & 3;                 // batch innermost: bias tiles reused in L2
  int h  = (bid >> 2) & 7;
  int ig = bid >> 5;                // 0..17
  int i0 = (ig * 8 + warp) * 16;
  int lj = lane & 15, sel = lane >> 4, msel8 = sel * 8;

  const _Float16* kbase = Kd + ((size_t)b * HEADS + h) * N_TOK * DHEAD;
  const _Float16* vbase = Vt + ((size_t)b * HEADS + h) * DHEAD * N_TOK;

  // per-thread staging source: waves 0-3 stage K chunk, waves 4-7 stage V
  bool isK = (t < 128);
  int  tt  = isK ? t : (t - 128);
  const _Float16* gsrc0 = isK
      ? (kbase + (size_t)(tt >> 2) * DHEAD + (tt & 3) * 8)   // K[key][d]
      : (vbase + (size_t)(tt >> 2) * N_TOK + (tt & 3) * 8);  // Vt[d][key]
  size_t   gstep  = isK ? (size_t)DHEAD : (size_t)1;         // per key index
  uint32_t lds0   = (uint32_t)(uintptr_t)(&KVs[0][0][0]) +
                    (isK ? 0u : 2048u) + (uint32_t)tt * 16u;

  auto stage = [&](int jc) {
    const void* g = (const void*)(gsrc0 + (size_t)(jc * 32) * gstep);
    uint32_t l = lds0 + (uint32_t)(jc & 1) * 4096u;
    asm volatile("global_load_async_to_lds_b128 %0, %1, off"
                 :: "v"(l), "v"(g) : "memory");
  };

  // Q A-fragment, held across all chunks
  const _Float16* qrow = Q + (((size_t)b * HEADS + h) * N_TOK + i0 + lj) * DHEAD;
  v16h qa = combine(*(const v8h*)(qrow + sel * 16),
                    *(const v8h*)(qrow + 8 + sel * 16));

  float Mx[8], Ls[8];
  v8f O0 = {}, O1 = {};
  #pragma unroll
  for (int r = 0; r < 8; r++) { Mx[r] = -3.0e38f; Ls[r] = 0.0f; }

  const float* bbase = bias + ((size_t)h * N_TOK + i0 + msel8) * N_TOK + lj;
  _Float16* pl = &Plds[warp][0];

  stage(0);
  asm volatile("s_wait_asynccnt 0x0" ::: "memory");
  __syncthreads();

  for (int jc = 0; jc < N_TOK / 32; jc++) {
    int j0 = jc * 32;
    if (jc + 1 < N_TOK / 32) stage(jc + 1);   // prefetch next chunk

    const _Float16* kb = &KVs[jc & 1][0][0];  // [32 keys][32 d]
    const _Float16* vb = &KVs[jc & 1][1][0];  // [32 d][32 keys]

    // S = Q K^T : K B-fragments from LDS (contiguous 32B per lane)
    const _Float16* kr0 = kb + (size_t)lj * 32 + sel * 16;
    const _Float16* kr1 = kb + (size_t)(lj + 16) * 32 + sel * 16;
    v16h kf0 = combine(*(const v8h*)kr0, *(const v8h*)(kr0 + 8));
    v16h kf1 = combine(*(const v8h*)kr1, *(const v8h*)(kr1 + 8));
    v8f s0 = {}, s1 = {};
    s0 = __builtin_amdgcn_wmma_f32_16x16x32_f16(false, qa, false, kf0,
                                                (short)0, s0, false, false);
    s1 = __builtin_amdgcn_wmma_f32_16x16x32_f16(false, qa, false, kf1,
                                                (short)0, s1, false, false);

    // bias add + online softmax
    float p0[8], p1[8], rmax[8];
    #pragma unroll
    for (int r = 0; r < 8; r++) {
      const float* bp = bbase + (size_t)r * N_TOK + j0;
      float v0 = s0[r] + bp[0];
      float v1 = s1[r] + bp[16];
      p0[r] = v0; p1[r] = v1;
      rmax[r] = fmaxf(v0, v1);
    }
    #pragma unroll
    for (int off = 1; off < 16; off <<= 1) {
      #pragma unroll
      for (int r = 0; r < 8; r++)
        rmax[r] = fmaxf(rmax[r], __shfl_xor(rmax[r], off, 32));
    }
    float corr[8];
    #pragma unroll
    for (int r = 0; r < 8; r++) {
      float mn = fmaxf(Mx[r], rmax[r]);
      corr[r] = __expf(Mx[r] - mn);
      Mx[r] = mn;
      p0[r] = __expf(p0[r] - mn);
      p1[r] = __expf(p1[r] - mn);
    }
    float rs[8];
    #pragma unroll
    for (int r = 0; r < 8; r++) rs[r] = p0[r] + p1[r];
    #pragma unroll
    for (int off = 1; off < 16; off <<= 1) {
      #pragma unroll
      for (int r = 0; r < 8; r++)
        rs[r] += __shfl_xor(rs[r], off, 32);
    }
    #pragma unroll
    for (int r = 0; r < 8; r++) {
      Ls[r] = Ls[r] * corr[r] + rs[r];
      O0[r] *= corr[r];
      O1[r] *= corr[r];
    }

    // P: C-layout -> per-wave LDS -> A-layout (DS ops in-order within a wave)
    #pragma unroll
    for (int r = 0; r < 8; r++) {
      pl[(r + msel8) * 32 + lj]      = (_Float16)p0[r];
      pl[(r + msel8) * 32 + lj + 16] = (_Float16)p1[r];
    }
    asm volatile("" ::: "memory");
    const _Float16* pr = pl + lj * 32 + sel * 16;
    v16h pa = combine(*(const v8h*)pr, *(const v8h*)(pr + 8));
    asm volatile("" ::: "memory");

    // O += P V : V B-fragments from LDS
    const _Float16* vr0 = vb + (size_t)lj * 32 + sel * 16;
    const _Float16* vr1 = vb + (size_t)(16 + lj) * 32 + sel * 16;
    v16h vf0 = combine(*(const v8h*)vr0, *(const v8h*)(vr0 + 8));
    v16h vf1 = combine(*(const v8h*)vr1, *(const v8h*)(vr1 + 8));
    O0 = __builtin_amdgcn_wmma_f32_16x16x32_f16(false, pa, false, vf0,
                                                (short)0, O0, false, false);
    O1 = __builtin_amdgcn_wmma_f32_16x16x32_f16(false, pa, false, vf1,
                                                (short)0, O1, false, false);

    // next chunk staged + everyone done reading current buffer
    asm volatile("s_wait_asynccnt 0x0" ::: "memory");
    __syncthreads();
  }

  #pragma unroll
  for (int r = 0; r < 8; r++) {
    float inv = 1.0f / Ls[r];
    size_t row = (size_t)b * N_TOK + i0 + r + msel8;
    AO[row * HIDDEN + h * DHEAD + lj]      = (_Float16)(O0[r] * inv);
    AO[row * HIDDEN + h * DHEAD + 16 + lj] = (_Float16)(O1[r] * inv);
  }
}

// ---------------------------------------------------------------------------
// Kernel 3: out = w_out @ AO + b_out (256 x 2304, K=256 per batch), f32 out.
// ---------------------------------------------------------------------------
__global__ __launch_bounds__(256) void proj_k(
    const _Float16* __restrict__ AO,     // [B*N][HIDDEN]
    const _Float16* __restrict__ wouth,  // [256][256] f16
    const float* __restrict__ bout,
    float* __restrict__ out)             // [B][256][N]
{
  int wid  = blockIdx.x * 8 + (threadIdx.x >> 5);
  int lane = threadIdx.x & 31;
  int jt = wid % (N_TOK / 16);
  int mt = (wid / (N_TOK / 16)) % (C_IN / 16);
  int b  = wid / ((N_TOK / 16) * (C_IN / 16));
  int m0 = mt * 16, j0 = jt * 16;
  int lj = lane & 15, sel = lane >> 4, msel8 = sel * 8;

  v8f acc = {};
  const _Float16* arow = wouth + (size_t)(m0 + lj) * HIDDEN;
  const _Float16* brow = AO + ((size_t)b * N_TOK + j0 + lj) * HIDDEN;
  #pragma unroll
  for (int kc = 0; kc < 8; kc++) {
    int k0 = kc * 32;
    const _Float16* pa = arow + k0 + sel * 16;
    v16h a = combine(*(const v8h*)pa, *(const v8h*)(pa + 8));
    const _Float16* pb = brow + k0 + sel * 16;
    v16h bf = combine(*(const v8h*)pb, *(const v8h*)(pb + 8));
    acc = __builtin_amdgcn_wmma_f32_16x16x32_f16(false, a, false, bf,
                                                 (short)0, acc, false, false);
  }
  #pragma unroll
  for (int r = 0; r < 8; r++) {
    int o = m0 + r + msel8;
    out[((size_t)b * C_IN + o) * N_TOK + j0 + lj] = acc[r] + bout[o];
  }
}

extern "C" void kernel_launch(void* const* d_in, const int* in_sizes, int n_in,
                              void* d_out, int out_size, void* d_ws, size_t ws_size,
                              hipStream_t stream) {
  const float* x        = (const float*)d_in[0];
  const float* pos_bias = (const float*)d_in[1];
  const float* w_qkv    = (const float*)d_in[2];
  const float* w_out    = (const float*)d_in[3];
  const float* b_out    = (const float*)d_in[4];
  float* out = (float*)d_out;

  size_t elems = (size_t)BATCH * HEADS * N_TOK * DHEAD;   // 2,359,296
  _Float16* Q     = (_Float16*)d_ws;
  _Float16* Kd    = Q + elems;
  _Float16* Vt    = Kd + elems;
  _Float16* AO    = Vt + elems;            // [B*N][HIDDEN]
  _Float16* xT    = AO + elems;            // [B][N][C]
  _Float16* wqkvh = xT + elems;            // 768*256
  _Float16* wouth = wqkvh + (size_t)O3 * C_IN;

  cvt_w_k  <<<dim3(768),  dim3(256), 0, stream>>>(w_qkv, w_out, wqkvh, wouth);
  xpose_k  <<<dim3(2304), dim3(256), 0, stream>>>(x, xT);
  qkv_gemm_k<<<dim3(3456), dim3(256), 0, stream>>>(xT, wqkvh, Q, Kd, Vt);
  attn_k   <<<dim3(576),  dim3(256), 0, stream>>>(Q, Kd, Vt, pos_bias, AO);
  proj_k   <<<dim3(1152), dim3(256), 0, stream>>>(AO, wouth, b_out, out);
}
